// HopfieldNetwork_23948737642912
// MI455X (gfx1250) — matrix operand order repaired and checked by reading, compiled, not run on hardware
//
#include <hip/hip_runtime.h>
#include <hip/hip_bf16.h>

#define B_BATCH 8
#define N_NEUR  4096
#define SWEEPS  10

typedef __attribute__((ext_vector_type(16))) _Float16 v16h;
typedef __attribute__((ext_vector_type(8)))  float    v8f;
typedef __attribute__((ext_vector_type(4)))  float    vf4;

// ---------------------------------------------------------------------------
// Kernel A: initial field H[b][n] = sum_k W[n][k] * x[b][k]
// WMMA f32_16x16x32_f16 with hi/lo split of W for near-f32 accuracy.
// One wave per 16-row tile of W; batch (8, duplicated to 16 cols) is the
// N dimension. Lanes 8..15 compute duplicate columns (never stored) so the
// inner loop has no exec-mask divergence around the WMMA operands.
// ---------------------------------------------------------------------------
__global__ __launch_bounds__(128) void hopfield_field_wmma(
    const float* __restrict__ W, const float* __restrict__ x,
    float* __restrict__ H)
{
    const int lane  = threadIdx.x & 31;
    const int warp  = threadIdx.x >> 5;
    const int tile  = blockIdx.x * 4 + warp;      // 256 tiles of 16 rows
    const int mrow  = lane & 15;                  // A-matrix row within tile
    const int khalf = lane >> 4;                  // 0 or 1 (K-half split)
    const int bcol  = lane & 15;                  // B/C column = batch index

    const int nrow_base = tile * 16;
    const float* Wr = W + (size_t)(nrow_base + mrow) * N_NEUR;
    const float* Xr = x + (size_t)(bcol & (B_BATCH - 1)) * N_NEUR; // dup cols 8..15

    v8f c = {};
    for (int k0 = 0; k0 < N_NEUR; k0 += 32) {
        // ---- A tile (16x32 f16), hi/lo split.
        // Lane covers K = khalf*16 + g*8 + [0..7] for g = 0,1 (contiguous, matches B)
        v16h ahi, alo;
        #pragma unroll
        for (int g = 0; g < 2; ++g) {
            const float* pa = Wr + k0 + khalf * 16 + g * 8;
            vf4 w0 = *(const vf4*)(pa);
            vf4 w1 = *(const vf4*)(pa + 4);
            float wv[8] = {w0.x, w0.y, w0.z, w0.w, w1.x, w1.y, w1.z, w1.w};
            #pragma unroll
            for (int j = 0; j < 8; ++j) {
                _Float16 hi = (_Float16)wv[j];
                ahi[g * 8 + j] = hi;
                alo[g * 8 + j] = (_Float16)(wv[j] - (float)hi);
            }
        }
        // ---- B tile (32x16 f16): lane covers K = khalf*16 + [0..15], column bcol
        v16h bm;
        {
            const float* pbp = Xr + k0 + khalf * 16;
            #pragma unroll
            for (int q = 0; q < 4; ++q) {
                vf4 xv = *(const vf4*)(pbp + q * 4);
                bm[q * 4 + 0] = (_Float16)xv.x;
                bm[q * 4 + 1] = (_Float16)xv.y;
                bm[q * 4 + 2] = (_Float16)xv.z;
                bm[q * 4 + 3] = (_Float16)xv.w;
            }
        }
        // D = A_lo*B + (A_hi*B + C)
        c = __builtin_amdgcn_wmma_f32_16x16x32_f16(false, ahi, false, bm,
                                                   (short)0, c, false, false);
        c = __builtin_amdgcn_wmma_f32_16x16x32_f16(false, alo, false, bm,
                                                   (short)0, c, false, false);
    }

    // C layout: VGPR v holds row M = v + 8*khalf, column bcol
    if (bcol < B_BATCH) {
        #pragma unroll
        for (int v = 0; v < 8; ++v) {
            const int nrow = nrow_base + v + 8 * khalf;
            H[(size_t)bcol * N_NEUR + nrow] = c[v];
        }
    }
}

// ---------------------------------------------------------------------------
// Kernel B: asynchronous sweeps via incremental field updates.
// One block per sample; h,y resident in LDS. On a sign flip at neuron idx:
//   h += delta * W[idx,:]   (parallel 16KB AXPY, no reduction)
// Next row is prefetched (global_prefetch_b8) one step ahead; the next
// permutation index is carried in a register so the serial critical path
// has a single scalar load per step.
// ---------------------------------------------------------------------------
__global__ __launch_bounds__(256) void hopfield_async_sweep(
    const float* __restrict__ W, const float* __restrict__ x,
    const int* __restrict__ perms, const float* __restrict__ H,
    float* __restrict__ out)
{
    __shared__ float h_s[N_NEUR];
    __shared__ float y_s[N_NEUR];

    const int b   = blockIdx.x;
    const int tid = threadIdx.x;
    const int TOT = SWEEPS * N_NEUR;

    for (int j = tid; j < N_NEUR; j += 256) {
        h_s[j] = H[(size_t)b * N_NEUR + j];
        y_s[j] = x[(size_t)b * N_NEUR + j];   // initial state is the query
    }
    __syncthreads();

    const int* pb = perms + (size_t)b * TOT;
    const int  j0 = tid * 16;                 // 16 consecutive floats per thread

    int idx = pb[0];
    for (int s = 0; s < TOT; ++s) {
        int nidx = idx;
        if (s + 1 < TOT) {
            nidx = pb[s + 1];
            __builtin_prefetch(W + (size_t)nidx * N_NEUR + j0, 0, 3);
        }
        const float hv = h_s[idx];            // LDS broadcast
        const float yo = y_s[idx];
        const float yn = (hv > 0.0f) ? 1.0f : ((hv < 0.0f) ? -1.0f : 0.0f);
        const float d  = yn - yo;
        if (d != 0.0f) {                      // uniform across the block
            __syncthreads();                  // all reads of h_s/y_s[idx] done
            const float4* wr = (const float4*)(W + (size_t)idx * N_NEUR + j0);
            float4 w0 = wr[0], w1 = wr[1], w2 = wr[2], w3 = wr[3];
            float* hp = h_s + j0;
            hp[ 0] += d * w0.x;  hp[ 1] += d * w0.y;
            hp[ 2] += d * w0.z;  hp[ 3] += d * w0.w;
            hp[ 4] += d * w1.x;  hp[ 5] += d * w1.y;
            hp[ 6] += d * w1.z;  hp[ 7] += d * w1.w;
            hp[ 8] += d * w2.x;  hp[ 9] += d * w2.y;
            hp[10] += d * w2.z;  hp[11] += d * w2.w;
            hp[12] += d * w3.x;  hp[13] += d * w3.y;
            hp[14] += d * w3.z;  hp[15] += d * w3.w;
            if (tid == 0) y_s[idx] = yn;
            __syncthreads();                  // updates visible to next step
        }
        idx = nidx;
    }

    __syncthreads();
    for (int j = tid; j < N_NEUR; j += 256)
        out[(size_t)b * N_NEUR + j] = y_s[j];
}

extern "C" void kernel_launch(void* const* d_in, const int* in_sizes, int n_in,
                              void* d_out, int out_size, void* d_ws, size_t ws_size,
                              hipStream_t stream) {
    const float* x     = (const float*)d_in[0];   // [B, N]  f32 (bipolar)
    const float* W     = (const float*)d_in[1];   // [N, N]  f32
    const int*   perms = (const int*)d_in[2];     // [B, ITERS, N] i32
    float*       out   = (float*)d_out;           // [B, N]  f32
    float*       H     = (float*)d_ws;            // [B, N]  f32 scratch (128 KB)

    // Phase 1: H = W * x^T via WMMA (256 tiles, 4 waves/block)
    hopfield_field_wmma<<<dim3((N_NEUR / 16) / 4), dim3(128), 0, stream>>>(W, x, H);
    // Phase 2: sequential async sweeps, one block per sample
    hopfield_async_sweep<<<dim3(B_BATCH), dim3(256), 0, stream>>>(W, x, perms, H, out);
}
